// VQAutoencoder_9706626089942
// MI455X (gfx1250) — matrix operand order repaired and checked by reading, not compiled, not run on hardware
//
#include <hip/hip_runtime.h>
#include <hip/hip_bf16.h>
#include <math.h>

// ---------------------------------------------------------------------------
// CDNA5 (gfx1250) types / feature detection
// ---------------------------------------------------------------------------
typedef __attribute__((ext_vector_type(16))) __bf16 v16bf;
typedef __attribute__((ext_vector_type(8)))  float  v8f;
typedef __attribute__((ext_vector_type(4)))  unsigned int u32x4;
typedef __attribute__((ext_vector_type(8)))  int    i32x8;
typedef __attribute__((ext_vector_type(4)))  int    i32x4;

#if __has_builtin(__builtin_amdgcn_tensor_load_to_lds) && __has_builtin(__builtin_amdgcn_s_wait_tensorcnt)
#define USE_TDM 1
#else
#define USE_TDM 0
#endif

// ---------------------------------------------------------------------------
// Implicit-GEMM conv: Out[co,p] = sum_k W[co,k] * im2col[k,p] (+bias)(+res)
// Block tile: 64 cout x 128 pixels, K-step 64.
// 8 waves as 2(m) x 4(n); each wave owns a 2x2 grid of 16x16 accumulators
// -> 8 WMMAs per wave per stage, ~22 FLOP per L2 byte.
// A (weights) staged via Tensor Data Mover (64 rows x 64 dwords, 1-dword pad
// per row -> LDS stride 65). B: hoisted im2col gather, one k-row and 32
// consecutive pixels per thread (OW, OHW % 32 == 0 throughout this net).
// ---------------------------------------------------------------------------
__global__ __launch_bounds__(256)
void k_conv_wmma(const float* __restrict__ in, const float* __restrict__ wgt,
                 const float* __restrict__ bias, const float* __restrict__ res,
                 float* __restrict__ out,
                 int B, int Cin, int IH, int IW, int Cout, int OH, int OW,
                 int KH, int KW, int stride, int padT, int padL)
{
    __shared__ float la[64 * 65];    // cout x k   (fp32 staged, stride 65)
    __shared__ float lb[64 * 129];   // k    x pix (fp32 staged, stride 129)

    const int K   = Cin * KH * KW;
    const int KHW = KH * KW;
    const int OHW = OH * OW;
    const int NP  = B * OHW;

    const int n0 = blockIdx.x * 128;  // pixel tile base
    const int m0 = blockIdx.y * 64;   // cout  tile base

    const int tid  = threadIdx.x;
    const int lane = tid & 31;
    const int wv   = tid >> 5;
    const int mwv  = (wv & 1) * 32;   // wave cout offset (2 waves in m)
    const int nwv  = (wv >> 1) * 32;  // wave pixel offset (4 waves in n)

    // B-gather ownership: one k-row, 32 consecutive pixels
    const int bkl  = tid >> 2;        // 0..63
    const int bnl0 = (tid & 3) * 32;  // 0,32,64,96
    // pixel-run decomposition is k0-invariant: hoist it
    const int nrun = n0 + bnl0;
    int run_b = 0, run_oy = 0, run_ox = 0, run_ok = 0;
    if (nrun < NP) {
        run_b  = nrun / OHW;
        int pos = nrun - run_b * OHW;
        run_oy = pos / OW;
        run_ox = pos - run_oy * OW;
        run_ok = 1;
    }
#if !USE_TDM
    // A-load ownership: one cout-row, 16 consecutive k
    const int aml  = tid >> 2;        // 0..63
    const int akl0 = (tid & 3) * 16;  // 0..48
#endif

    v8f acc[2][2] = {};

    for (int k0 = 0; k0 < K; k0 += 64) {
#if USE_TDM
        // --- A tile via TDM: 2D tile 64 x 64 dwords, 1-dword pad per row ---
        if (wv == 0) {
            unsigned ldsoff = (unsigned)(unsigned long long)(&la[0]);
            unsigned long long ga =
                (unsigned long long)(const void*)(wgt + (size_t)m0 * K + k0);
            int rem_k = K - k0;     if (rem_k < 0) rem_k = 0;
            int rem_m = Cout - m0;  if (rem_m < 0) rem_m = 0;
            u32x4 g0;
            g0[0] = 1u;                                    // count=1, user mode
            g0[1] = ldsoff;                                // lds_addr
            g0[2] = (unsigned)ga;                          // global_addr lo
            g0[3] = (unsigned)((ga >> 32) & 0x1FFFFFFu) | (2u << 30); // hi|type=2
            i32x8 g1;
            // data_size=4B(2), pad_enable, pad_interval=64dw(enc 5), pad=1dw(enc 0)
            g1[0] = (int)((2u << 16) | (1u << 20) | (5u << 22));
            g1[1] = (int)(((unsigned)rem_k & 0xFFFFu) << 16);          // dim0 lo16
            g1[2] = (int)((((unsigned)rem_k >> 16) & 0xFFFFu) |
                          (((unsigned)rem_m & 0xFFFFu) << 16));        // dim0 hi | dim1 lo
            g1[3] = (int)((((unsigned)rem_m >> 16) & 0xFFFFu) |
                          (64u << 16));                                // dim1 hi | tile_dim0
            g1[4] = 64;                                    // tile_dim1=64, tile_dim2=0
            g1[5] = K;                                     // tensor_dim0_stride lo32
            g1[6] = 0;
            g1[7] = 0;
            i32x4 z4 = {0, 0, 0, 0};
            i32x8 z8 = {0, 0, 0, 0, 0, 0, 0, 0};
            __builtin_amdgcn_tensor_load_to_lds(g0, g1, z4, z4, z8, 0);
            if (tid == 0) __builtin_prefetch(wgt + (size_t)m0 * K + k0 + 64, 0, 0);
        }
#else
        // --- A tile fallback: contiguous 16-float run per thread ---
        {
            int m = m0 + aml;
            const float* wr = wgt + (size_t)m * K + k0 + akl0;
            float* dst = &la[aml * 65 + akl0];
            #pragma unroll
            for (int e = 0; e < 16; e++) {
                int k = k0 + akl0 + e;
                dst[e] = (m < Cout && k < K) ? wr[e] : 0.f;
            }
        }
#endif
        // --- B tile: hoisted im2col gather, 32 contiguous pixels per thread ---
        {
            float* dst = &lb[bkl * 129 + bnl0];
            const int k = k0 + bkl;
            if (run_ok && k < K) {
                int ci = k / KHW;
                int r  = k - ci * KHW;
                int ky = r / KW;
                int kx = r - ky * KW;
                int iy = run_oy * stride + ky - padT;
                if ((unsigned)iy < (unsigned)IH) {
                    const float* src =
                        in + (((size_t)run_b * Cin + ci) * IH + iy) * IW;
                    int ixb = run_ox * stride + kx - padL;
                    #pragma unroll
                    for (int e = 0; e < 32; e++) {
                        int ix = ixb + e * stride;
                        dst[e] = ((unsigned)ix < (unsigned)IW) ? src[ix] : 0.f;
                    }
                } else {
                    #pragma unroll
                    for (int e = 0; e < 32; e++) dst[e] = 0.f;
                }
            } else {
                #pragma unroll
                for (int e = 0; e < 32; e++) dst[e] = 0.f;
            }
        }
#if USE_TDM
        if (wv == 0) __builtin_amdgcn_s_wait_tensorcnt(0);
#endif
        __syncthreads();

        // --- 2 k-halves x 2x2 register block = 8 WMMAs per wave per stage ---
        {
            const int mm = lane & 15, hi = lane >> 4, ks = hi * 8;
            #pragma unroll
            for (int h2 = 0; h2 < 2; h2++) {
                const int koff = h2 * 32;
                v16bf af[2], bfm[2];
                #pragma unroll
                for (int ms = 0; ms < 2; ms++) {
                    const int row = (mwv + ms * 16 + mm) * 65 + koff;
                    #pragma unroll
                    for (int j = 0; j < 8; j++)
                        af[ms][j] = (__bf16)la[row + ks + j];
                    #pragma unroll
                    for (int j = 0; j < 8; j++)
                        af[ms][8 + j] = (__bf16)la[row + 16 + ks + j];
                }
                #pragma unroll
                for (int ns = 0; ns < 2; ns++) {
                    const int col = nwv + ns * 16 + mm;
                    #pragma unroll
                    for (int j = 0; j < 16; j++)
                        bfm[ns][j] = (__bf16)lb[(koff + hi * 8 + j + (j >> 3) * 8) * 129 + col];
                }
                #pragma unroll
                for (int ms = 0; ms < 2; ms++)
                    #pragma unroll
                    for (int ns = 0; ns < 2; ns++)
                        acc[ms][ns] = __builtin_amdgcn_wmma_f32_16x16x32_bf16(
                            false, af[ms], false, bfm[ns], (short)0,
                            acc[ms][ns], false, false);
            }
        }
        __syncthreads();
    }

    // --- epilogue: D layout (VGPR r: M=r / r+8, N=lane), bias + residual ---
    const int mm = lane & 15, hi = lane >> 4;
    #pragma unroll
    for (int ns = 0; ns < 2; ns++) {
        const int n = n0 + nwv + ns * 16 + mm;
        if (n < NP) {
            const int b   = n / OHW;
            const int pos = n - b * OHW;
            float*       op = out + (size_t)b * Cout * OHW + pos;
            const float* rp = res ? res + (size_t)b * Cout * OHW + pos : nullptr;
            #pragma unroll
            for (int ms = 0; ms < 2; ms++) {
                #pragma unroll
                for (int j = 0; j < 8; j++) {
                    int m = m0 + mwv + ms * 16 + hi * 8 + j;
                    if (m < Cout) {
                        float v = acc[ms][ns][j] + bias[m];
                        if (rp) v += rp[(size_t)m * OHW];
                        op[(size_t)m * OHW] = v;
                    }
                }
            }
        }
    }
}

// ---------------------------------------------------------------------------
// Generic strided WMMA GEMM: C[m,n] = alpha * sum_k A[m,k]*B[k,n]
// (attention QK^T and A*V). Same 64x128 block tile / 2x2 register block.
// ---------------------------------------------------------------------------
__global__ __launch_bounds__(256)
void k_gemm_wmma(const float* __restrict__ A, const float* __restrict__ Bm,
                 float* __restrict__ Cm, int M, int N, int K,
                 int sam, int sak, int sbk, int sbn, int scm, int scn,
                 float alpha)
{
    __shared__ float la[64 * 65];
    __shared__ float lb[64 * 129];

    const int n0 = blockIdx.x * 128;
    const int m0 = blockIdx.y * 64;
    const int tid = threadIdx.x, lane = tid & 31, wv = tid >> 5;
    const int mwv = (wv & 1) * 32, nwv = (wv >> 1) * 32;

    const int aml  = tid >> 2;        // A: one m-row, 16 consecutive k
    const int akl0 = (tid & 3) * 16;
    const int bkl  = tid >> 2;        // B: one k-row, 32 consecutive n
    const int bnl0 = (tid & 3) * 32;

    v8f acc[2][2] = {};

    for (int k0 = 0; k0 < K; k0 += 64) {
        {
            int m = m0 + aml;
            const float* ap = A + (size_t)m * sam + (size_t)(k0 + akl0) * sak;
            float* dst = &la[aml * 65 + akl0];
            #pragma unroll
            for (int e = 0; e < 16; e++) {
                int k = k0 + akl0 + e;
                dst[e] = (m < M && k < K) ? ap[(size_t)e * sak] : 0.f;
            }
        }
        {
            int k = k0 + bkl;
            const float* bp = Bm + (size_t)k * sbk + (size_t)(n0 + bnl0) * sbn;
            float* dst = &lb[bkl * 129 + bnl0];
            #pragma unroll
            for (int e = 0; e < 32; e++) {
                int n = n0 + bnl0 + e;
                dst[e] = (k < K && n < N) ? bp[(size_t)e * sbn] : 0.f;
            }
        }
        __syncthreads();
        {
            const int mm = lane & 15, hi = lane >> 4, ks = hi * 8;
            #pragma unroll
            for (int h2 = 0; h2 < 2; h2++) {
                const int koff = h2 * 32;
                v16bf af[2], bfm[2];
                #pragma unroll
                for (int ms = 0; ms < 2; ms++) {
                    const int row = (mwv + ms * 16 + mm) * 65 + koff;
                    #pragma unroll
                    for (int j = 0; j < 8; j++)
                        af[ms][j] = (__bf16)la[row + ks + j];
                    #pragma unroll
                    for (int j = 0; j < 8; j++)
                        af[ms][8 + j] = (__bf16)la[row + 16 + ks + j];
                }
                #pragma unroll
                for (int ns = 0; ns < 2; ns++) {
                    const int col = nwv + ns * 16 + mm;
                    #pragma unroll
                    for (int j = 0; j < 16; j++)
                        bfm[ns][j] = (__bf16)lb[(koff + hi * 8 + j + (j >> 3) * 8) * 129 + col];
                }
                #pragma unroll
                for (int ms = 0; ms < 2; ms++)
                    #pragma unroll
                    for (int ns = 0; ns < 2; ns++)
                        acc[ms][ns] = __builtin_amdgcn_wmma_f32_16x16x32_bf16(
                            false, af[ms], false, bfm[ns], (short)0,
                            acc[ms][ns], false, false);
            }
        }
        __syncthreads();
    }

    const int mm = lane & 15, hi = lane >> 4;
    #pragma unroll
    for (int ns = 0; ns < 2; ns++) {
        const int n = n0 + nwv + ns * 16 + mm;
        if (n < N) {
            #pragma unroll
            for (int ms = 0; ms < 2; ms++) {
                #pragma unroll
                for (int j = 0; j < 8; j++) {
                    int m = m0 + mwv + ms * 16 + hi * 8 + j;
                    if (m < M)
                        Cm[(size_t)m * scm + (size_t)n * scn] =
                            alpha * acc[ms][ns][j];
                }
            }
        }
    }
}

// ---------------------------------------------------------------------------
// GroupNorm(32) + optional SiLU. One block per (batch, group).
// ---------------------------------------------------------------------------
__global__ __launch_bounds__(256)
void k_gn_silu(const float* __restrict__ in, const float* __restrict__ g,
               const float* __restrict__ b, float* __restrict__ out,
               int C, int H, int W, int silu)
{
    const int bg    = blockIdx.x;
    const int batch = bg >> 5;
    const int grp   = bg & 31;
    const int cg    = C >> 5;
    const int HW    = H * W;
    const int n     = cg * HW;
    const float* base = in + ((size_t)batch * C + (size_t)grp * cg) * HW;
    float*       ob   = out + ((size_t)batch * C + (size_t)grp * cg) * HW;

    float s = 0.f, s2 = 0.f;
    for (int i = threadIdx.x; i < n; i += 256) {
        float v = base[i]; s += v; s2 += v * v;
    }
    __shared__ float rs[256], rq[256];
    rs[threadIdx.x] = s; rq[threadIdx.x] = s2;
    __syncthreads();
    for (int o = 128; o > 0; o >>= 1) {
        if (threadIdx.x < o) { rs[threadIdx.x] += rs[threadIdx.x + o];
                               rq[threadIdx.x] += rq[threadIdx.x + o]; }
        __syncthreads();
    }
    const float mean = rs[0] / (float)n;
    const float var  = rq[0] / (float)n - mean * mean;
    const float inv  = rsqrtf(var + 1e-6f);

    for (int i = threadIdx.x; i < n; i += 256) {
        int c = grp * cg + i / HW;
        float v = (base[i] - mean) * inv * g[c] + b[c];
        if (silu) v = v / (1.f + __expf(-v));
        ob[i] = v;
    }
}

// ---------------------------------------------------------------------------
__global__ __launch_bounds__(256)
void k_softmax_rows(float* __restrict__ S, int N)
{
    float* row = S + (size_t)blockIdx.x * N;
    __shared__ float sh[256];
    float mx = -1e30f;
    for (int i = threadIdx.x; i < N; i += 256) mx = fmaxf(mx, row[i]);
    sh[threadIdx.x] = mx; __syncthreads();
    for (int o = 128; o > 0; o >>= 1) {
        if (threadIdx.x < o) sh[threadIdx.x] = fmaxf(sh[threadIdx.x], sh[threadIdx.x + o]);
        __syncthreads();
    }
    mx = sh[0]; __syncthreads();
    float s = 0.f;
    for (int i = threadIdx.x; i < N; i += 256) {
        float e = __expf(row[i] - mx); row[i] = e; s += e;
    }
    sh[threadIdx.x] = s; __syncthreads();
    for (int o = 128; o > 0; o >>= 1) {
        if (threadIdx.x < o) sh[threadIdx.x] += sh[threadIdx.x + o];
        __syncthreads();
    }
    float inv = 1.f / sh[0];
    for (int i = threadIdx.x; i < N; i += 256) row[i] *= inv;
}

// ---------------------------------------------------------------------------
__global__ __launch_bounds__(256)
void k_upsample2(const float* __restrict__ in, float* __restrict__ out,
                 int NC, int H, int W)
{
    const int oW = 2 * W, oH = 2 * H;
    const size_t tot = (size_t)NC * oH * oW;
    for (size_t i = (size_t)blockIdx.x * 256 + threadIdx.x; i < tot;
         i += (size_t)gridDim.x * 256) {
        int    ox = (int)(i % oW);
        size_t t  = i / oW;
        int    oy = (int)(t % oH);
        size_t n  = t / oH;
        out[i] = in[(n * H + (oy >> 1)) * W + (ox >> 1)];
    }
}

// ---------------------------------------------------------------------------
__global__ void k_zero(float* p, int n)
{
    int i = blockIdx.x * 256 + threadIdx.x;
    if (i < n) p[i] = 0.f;
}

// ---------------------------------------------------------------------------
// VQ: per position argmin over K=8192 codebook entries of dim 3.
// Writes q (value of zq), token ids (as float), counts histogram, SSE accum.
// ---------------------------------------------------------------------------
__global__ __launch_bounds__(256)
void k_vq(const float* __restrict__ z, const float* __restrict__ cb,
          float* __restrict__ q, float* __restrict__ tok,
          float* __restrict__ counts, float* __restrict__ accum)
{
    const int n   = blockIdx.x * 256 + threadIdx.x;   // 0..2047
    const int b   = n >> 10;
    const int pos = n & 1023;
    const float* zp = z + (size_t)b * 3 * 1024 + pos;
    const float f0 = zp[0], f1 = zp[1024], f2 = zp[2048];

    int best = 0; float bd = 1e30f;
    for (int k = 0; k < 8192; k++) {
        float d0 = f0 - cb[k * 3 + 0];
        float d1 = f1 - cb[k * 3 + 1];
        float d2 = f2 - cb[k * 3 + 2];
        float d = d0 * d0 + d1 * d1 + d2 * d2;
        if (d < bd) { bd = d; best = k; }
    }
    float* qp = q + (size_t)b * 3 * 1024 + pos;
    qp[0]    = cb[best * 3 + 0];
    qp[1024] = cb[best * 3 + 1];
    qp[2048] = cb[best * 3 + 2];
    tok[n] = (float)best;
    atomicAdd(&counts[best], 1.0f);

    __shared__ float sh[256];
    sh[threadIdx.x] = bd; __syncthreads();
    for (int o = 128; o > 0; o >>= 1) {
        if (threadIdx.x < o) sh[threadIdx.x] += sh[threadIdx.x + o];
        __syncthreads();
    }
    if (threadIdx.x == 0) atomicAdd(accum, sh[0]);
}

__global__ __launch_bounds__(256)
void k_finalize(const float* __restrict__ counts, const float* __restrict__ accum,
                float* __restrict__ loss, float* __restrict__ ppl)
{
    __shared__ float sh[256];
    float se = 0.f;
    for (int i = threadIdx.x; i < 8192; i += 256) {
        float p = counts[i] / 2048.0f;
        se += p * __logf(p + 1e-10f);
    }
    sh[threadIdx.x] = se; __syncthreads();
    for (int o = 128; o > 0; o >>= 1) {
        if (threadIdx.x < o) sh[threadIdx.x] += sh[threadIdx.x + o];
        __syncthreads();
    }
    if (threadIdx.x == 0) {
        *ppl  = __expf(-sh[0]);
        *loss = accum[0] * 1.25f / 6144.0f;   // (1+BETA)*mean||q-z||^2
    }
}

// ---------------------------------------------------------------------------
// Host orchestration
// ---------------------------------------------------------------------------
struct PP { const float* a; const float* b; };

extern "C" void kernel_launch(void* const* d_in, const int* in_sizes, int n_in,
                              void* d_out, int out_size, void* d_ws, size_t ws_size,
                              hipStream_t stream)
{
    (void)in_sizes; (void)out_size;
    const size_t BUF  = 33554432;                   // 2*256*128*128 floats
    const size_t NEED = 5 * BUF + 65536;
    if (ws_size < NEED || n_in < 10) return;

    char* wsb = (char*)d_ws;
    float* buf[5];
    for (int i = 0; i < 5; i++) buf[i] = (float*)(wsb + (size_t)i * BUF);
    float* counts = (float*)(wsb + 5 * BUF);
    float* accum  = counts + 8192;

    int cur = 0;
    auto nxt   = [&]() { return (const float*)d_in[cur++]; };
    auto pair2 = [&]() { PP p; p.a = nxt(); p.b = nxt(); return p; };

    const float* x = nxt();                         // input image [2,3,128,128]
    const int Bn = 2;

    const float* act = nullptr;
    int ia = 0, C = 0, H = 0, W = 0;

    auto conv_launch = [&](const float* in, PP p, const float* res, float* out,
                           int Cin, int IH, int IW, int Cout, int OH, int OW,
                           int ks, int stride, int padT, int padL) {
        int NP = Bn * OH * OW;
        dim3 g((NP + 127) / 128, (Cout + 63) / 64);
        k_conv_wmma<<<g, 256, 0, stream>>>(in, p.a, p.b, res, out,
                                           Bn, Cin, IH, IW, Cout, OH, OW,
                                           ks, ks, stride, padT, padL);
    };
    auto gemm_launch = [&](const float* A, const float* Bm, float* Cm,
                           int M, int N, int K, int sam, int sak,
                           int sbk, int sbn, int scm, int scn, float alpha) {
        dim3 g((N + 127) / 128, (M + 63) / 64);
        k_gemm_wmma<<<g, 256, 0, stream>>>(A, Bm, Cm, M, N, K,
                                           sam, sak, sbk, sbn, scm, scn, alpha);
    };
    auto gn_launch = [&](const float* in, PP p, float* out, int Cc, int silu) {
        k_gn_silu<<<dim3(Bn * 32), 256, 0, stream>>>(in, p.a, p.b, out, Cc, H, W, silu);
    };

    auto resblock = [&](int Cout) {
        PP gn1 = pair2(), c1 = pair2(), gn2 = pair2(), c2 = pair2();
        PP sc{nullptr, nullptr};
        bool has_sc = (C != Cout);
        if (has_sc) sc = pair2();
        float* t = buf[(ia + 1) % 5];
        float* h = buf[(ia + 2) % 5];
        float* s = buf[(ia + 3) % 5];
        float* o = buf[(ia + 4) % 5];
        gn_launch(act, gn1, t, C, 1);
        conv_launch(t, c1, nullptr, h, C, H, W, Cout, H, W, 3, 1, 1, 1);
        gn_launch(h, gn2, t, Cout, 1);
        const float* r = act;
        if (has_sc) { conv_launch(act, sc, nullptr, s, C, H, W, Cout, H, W, 1, 1, 0, 0); r = s; }
        conv_launch(t, c2, r, o, Cout, H, W, Cout, H, W, 3, 1, 1, 1);
        act = o; ia = (ia + 4) % 5; C = Cout;
    };

    auto attnblk = [&]() {
        PP gn = pair2(), qkv = pair2(), proj = pair2();
        const int HWl = H * W;
        float* t  = buf[(ia + 1) % 5];
        float* qb = buf[(ia + 2) % 5];
        float* Sb = buf[(ia + 3) % 5];
        float* ob = buf[(ia + 4) % 5];
        gn_launch(act, gn, t, C, 0);
        conv_launch(t, qkv, nullptr, qb, C, H, W, 3 * C, H, W, 1, 1, 0, 0);
        float alpha = 1.0f / sqrtf((float)C);       // (c^-0.25)^2 on both q,k
        for (int b = 0; b < Bn; b++) {
            const float* q  = qb + (size_t)b * 3 * C * HWl;
            const float* kk = q + (size_t)C * HWl;
            const float* v  = kk + (size_t)C * HWl;
            float* S = Sb + (size_t)b * HWl * HWl;
            // S[i,j] = alpha * sum_c q[c,i] k[c,j]
            gemm_launch(q, kk, S, HWl, HWl, C, 1, HWl, HWl, 1, HWl, 1, alpha);
            k_softmax_rows<<<HWl, 256, 0, stream>>>(S, HWl);
            // o[c,i] = sum_j v[c,j] * S[i,j]
            gemm_launch(v, S, ob + (size_t)b * C * HWl, C, HWl, HWl,
                        HWl, 1, 1, HWl, HWl, 1, 1.0f);
        }
        conv_launch(ob, proj, act, t, C, H, W, C, H, W, 1, 1, 0, 0);   // + x
        act = t; ia = (ia + 1) % 5;
    };

    // ===================== encoder =====================
    {
        PP c1 = pair2();
        C = 128; H = 128; W = 128;
        conv_launch(x, c1, nullptr, buf[0], 3, 128, 128, 128, 128, 128, 3, 1, 1, 1);
        act = buf[0]; ia = 0;
        const int dims[4] = {128, 128, 256, 512};
        for (int i = 0; i < 3; i++) {
            int dout = dims[i + 1];
            resblock(dout);
            resblock(dout);
            if (i != 2) {                           // ZeroPad2d((0,1,0,1)) + s2 conv
                PP dn = pair2();
                float* o = buf[(ia + 1) % 5];
                conv_launch(act, dn, nullptr, o, C, H, W, C, H / 2, W / 2, 3, 2, 0, 0);
                act = o; ia = (ia + 1) % 5; H /= 2; W /= 2;
            }
        }
        resblock(512);                              // mid0
        attnblk();                                  // mid1
        resblock(512);                              // mid2
        PP hgn = pair2(), hc = pair2();
        float* t  = buf[(ia + 1) % 5];
        float* zb = buf[(ia + 2) % 5];
        gn_launch(act, hgn, t, C, 1);
        conv_launch(t, hc, nullptr, zb, C, H, W, 3, H, W, 3, 1, 1, 1);
        act = zb; ia = (ia + 2) % 5; C = 3;
    }
    // quant conv (1x1 3->3)
    {
        PP qc = pair2();
        float* z = buf[(ia + 1) % 5];
        conv_launch(act, qc, nullptr, z, 3, H, W, 3, H, W, 1, 1, 0, 0);
        act = z; ia = (ia + 1) % 5;
    }
    // ===================== vector quantizer =====================
    float* yout   = (float*)d_out;
    float* tokens = yout + 98304;
    float* lossp  = tokens + 2048;
    float* pplp   = lossp + 1;
    {
        const float* cb = nxt();                    // codebook [8192,3]
        float* qbuf = buf[(ia + 1) % 5];
        k_zero<<<(8192 + 255) / 256, 256, 0, stream>>>(counts, 8192);
        k_zero<<<1, 256, 0, stream>>>(accum, 1);
        k_vq<<<8, 256, 0, stream>>>(act, cb, qbuf, tokens, counts, accum);
        k_finalize<<<1, 256, 0, stream>>>(counts, accum, lossp, pplp);
        act = qbuf; ia = (ia + 1) % 5;              // zq value == q
        PP pqc = pair2();                           // post-quant conv
        float* zq = buf[(ia + 1) % 5];
        conv_launch(act, pqc, nullptr, zq, 3, H, W, 3, H, W, 1, 1, 0, 0);
        act = zq; ia = (ia + 1) % 5;
    }
    // ===================== decoder =====================
    {
        PP dc1 = pair2();
        float* d0 = buf[(ia + 1) % 5];
        conv_launch(act, dc1, nullptr, d0, 3, H, W, 512, H, W, 3, 1, 1, 1);
        act = d0; ia = (ia + 1) % 5; C = 512;
        resblock(512); attnblk(); resblock(512);
        const int ddims[4] = {512, 512, 256, 128};
        for (int i = 0; i < 3; i++) {
            int dout = ddims[i + 1];
            resblock(dout);
            resblock(dout);
            resblock(dout);
            if (i != 2) {                           // nearest x2 + conv3 p1
                PP up = pair2();
                float* u = buf[(ia + 1) % 5];
                float* o = buf[(ia + 2) % 5];
                size_t tot = (size_t)Bn * C * 4 * H * W;
                k_upsample2<<<dim3((unsigned)((tot + 255) / 256)), 256, 0, stream>>>(
                    act, u, Bn * C, H, W);
                H *= 2; W *= 2;
                conv_launch(u, up, nullptr, o, C, H, W, C, H, W, 3, 1, 1, 1);
                act = o; ia = (ia + 2) % 5;
            }
        }
        PP hgn = pair2(), hc = pair2();
        float* t = buf[(ia + 1) % 5];
        gn_launch(act, hgn, t, C, 1);
        conv_launch(t, hc, nullptr, yout, C, H, W, 3, H, W, 3, 1, 1, 1);
    }
}
